// RNN1_27745488732754
// MI455X (gfx1250) — matrix operand order repaired and checked by reading, not compiled, not run on hardware
//
#include <hip/hip_runtime.h>
#include <cstdint>

// ---------------------------------------------------------------------------
// CDNA5 (gfx1250) bf16 WMMA types & helpers
// ---------------------------------------------------------------------------
typedef __bf16 bf16;
typedef __bf16 v16bf __attribute__((ext_vector_type(16)));
typedef float  v8f   __attribute__((ext_vector_type(8)));
typedef unsigned int u32x4 __attribute__((ext_vector_type(4)));
typedef int          i32x4 __attribute__((ext_vector_type(4)));
typedef int          i32x8 __attribute__((ext_vector_type(8)));

union ABf { v16bf v; bf16 h[16]; unsigned int w[8]; };

__device__ __forceinline__ bf16 tobf(float f) { return (bf16)f; }

__device__ __forceinline__ v8f wmma_bf16(v16bf a, v16bf b, v8f c) {
    // D = A(16x32 bf16) * B(32x16 bf16) + C(16x16 f32)
    return __builtin_amdgcn_wmma_f32_16x16x32_bf16(
        false, a, false, b, (short)0, c, false, false);
}

// A-fragment K offset for VGPR j / lane-half h (contiguous K: k = 16*half + 2*j, matches B)
__device__ __forceinline__ int a_k0(int j, int half) {
    return half * 16 + 2 * j;
}

// A fragment from bf16 row (contiguous K), offset kbase (even).
__device__ __forceinline__ v16bf make_a_frag(const bf16* rowbase, int kbase, int half) {
    ABf f;
#pragma unroll
    for (int j = 0; j < 8; ++j)
        f.w[j] = *(const unsigned int*)(rowbase + kbase + a_k0(j, half));
    return f.v;
}

// A fragment from f32 LDS row: ds_load_b64 pairs + v_cvt_pk_bf16_f32.
__device__ __forceinline__ v16bf make_a_frag_f32lds(const float* rowbase, int half) {
    ABf f;
#pragma unroll
    for (int j = 0; j < 8; ++j) {
        int k = a_k0(j, half);
        f.h[2 * j]     = tobf(rowbase[k]);
        f.h[2 * j + 1] = tobf(rowbase[k + 1]);
    }
    return f.v;
}

// B fragment from bf16 LDS; Wrow points at this column's contiguous K row.
__device__ __forceinline__ v16bf make_b_frag_lds(const bf16* Wrow, int kb) {
    ABf f;
#pragma unroll
    for (int j = 0; j < 8; ++j)
        f.w[j] = *(const unsigned int*)(Wrow + kb + 2 * j);
    return f.v;
}

// B fragment from f32 global weight row (unguarded fast path).
__device__ __forceinline__ v16bf make_b_frag_f32_fast(const float* __restrict__ wrow, int kb) {
    ABf f;
#pragma unroll
    for (int j = 0; j < 16; ++j) f.h[j] = tobf(wrow[kb + j]);
    return f.v;
}

// B fragment from f32 global weight row with K guard (tail chunk only).
__device__ __forceinline__ v16bf make_b_frag_f32_guard(const float* __restrict__ wrow,
                                                       int kb, int K) {
    ABf f;
#pragma unroll
    for (int j = 0; j < 16; ++j) {
        int kg = kb + j;
        f.h[j] = tobf(kg < K ? wrow[kg] : 0.f);
    }
    return f.v;
}

// ---------------------------------------------------------------------------
// Tensor Data Mover: 2-D f32 tile DMA global -> LDS (gfx1250 TDM D#, ISA 08)
// Reads past tensor_w return zero (free K-tail padding).
// ---------------------------------------------------------------------------
#if __has_builtin(__builtin_amdgcn_tensor_load_to_lds)
#define HAVE_TDM 1
#else
#define HAVE_TDM 0
#endif

#if HAVE_TDM
__device__ __forceinline__ void tdm_load_f32_2d(unsigned lds_byte_addr, const float* gsrc,
                                                unsigned tensor_w,  // valid elems per row
                                                unsigned tile_w,    // elems per row to move
                                                unsigned rows,      // tile rows
                                                unsigned row_stride /*elems*/) {
    unsigned long long ga = (unsigned long long)(uintptr_t)gsrc;
    u32x4 g0;
    g0[0] = 1u;                                          // count=1 (user D#)
    g0[1] = lds_byte_addr;                               // lds_addr
    g0[2] = (unsigned)ga;                                // global_addr[31:0]
    g0[3] = ((unsigned)(ga >> 32) & 0x01FFFFFFu)         // global_addr[56:32]
            | (2u << 30);                                // type = 2 ("image")
    i32x8 g1;
    g1[0] = 0x00020000;                                  // data_size = 4B
    g1[1] = (int)((tensor_w & 0xFFFFu) << 16);           // tensor_dim0[15:0]
    g1[2] = (int)((tensor_w >> 16) & 0xFFFFu)            // tensor_dim0[31:16]
          | (int)((rows & 0xFFFFu) << 16);               // tensor_dim1[15:0]
    g1[3] = (int)((rows >> 16) & 0xFFFFu)                // tensor_dim1[31:16]
          | (int)(tile_w << 16);                         // tile_dim0
    g1[4] = (int)rows;                                   // tile_dim1 (tile_dim2 = 0)
    g1[5] = (int)row_stride;                             // tensor_dim0_stride[31:0]
    g1[6] = 0;
    g1[7] = 0;
    i32x4 z4 = {0, 0, 0, 0};
#if __has_include(<hip/amd_detail/amd_gfx1250_TDM.h>)
    i32x8 z8 = {0, 0, 0, 0, 0, 0, 0, 0};                 // amdgpu-toolchain: 6-arg form
    __builtin_amdgcn_tensor_load_to_lds(g0, g1, z4, z4, z8, 0);
#else
    __builtin_amdgcn_tensor_load_to_lds(g0, g1, z4, z4, 0);  // ROCm 7.2: 5-arg form
#endif
}
#endif

// ---------------------------------------------------------------------------
// Problem constants
// ---------------------------------------------------------------------------
#define NT     64
#define BATCH  4096
#define ESZ    8385        // E_SHAPE
#define KFEAT  535         // 10 emb + 525 cont
#define KPAD   544         // 17 * 32

// ---------------------------------------------------------------------------
// Kernel 1: fused embedding-gather + concat + fc1.
// Steady state: TDM double-buffers 64x32 f32 X_cont tiles into LDS while
// WMMA consumes the previous tile; chunk 0 (embedding) and 16 (pad) are
// staged cooperatively. fc1_w lives in LDS as bf16.
// grid (64 batch-tiles of 64 rows, 64 n), block 128 (4 waves)
// ---------------------------------------------------------------------------
__global__ void fc1_kernel(const int* __restrict__ disc_idx,
                           const float* __restrict__ X_cont,
                           const float* __restrict__ embedding,
                           const float* __restrict__ fc1_w,
                           const float* __restrict__ fc1_b,
                           bf16* __restrict__ Xb) {
    __shared__ bf16  wbf[32 * KPAD];      // 34,816 B: fc1_w bf16, K zero-padded
    __shared__ float fstage[2][64 * 32];  // 2 x 8,192 B ping-pong f32 tiles
    const int tid  = threadIdx.x;
    const int lane = tid & 31, wave = tid >> 5;
    const int half = lane >> 4, nloc = lane & 15;
    const int n    = blockIdx.y;
    const int bt0  = blockIdx.x * 64;
    const float* Xrow0 = X_cont + ((size_t)n * BATCH + bt0) * 525;

    // one-time: fc1_w -> bf16 LDS (zero-pad K in [535,544))
    for (int p = tid; p < 32 * KPAD; p += 128) {
        int o = p / KPAD, k = p % KPAD;
        wbf[p] = tobf(k < KFEAT ? fc1_w[o * KFEAT + k] : 0.f);
    }
    // one-time: chunk 0 (embedding gather + first 22 cont cols) -> fstage[0]
    for (int p = tid; p < 64 * 32; p += 128) {
        int r = p >> 5, k = p & 31;
        float v;
        if (k < 10) {
            int e = disc_idx[n * BATCH + bt0 + r];
            v = embedding[e * 10 + k];
        } else {
            v = Xrow0[(size_t)r * 525 + (k - 10)];
        }
        fstage[0][p] = v;
    }

    v8f acc0 = {}, acc1 = {};
#pragma unroll 1
    for (int kk = 0; kk < 17; ++kk) {
        const int kbase = kk * 32;
        float* cur = fstage[kk & 1];
#if HAVE_TDM
        // issue DMA for chunk kk+1 (chunks 1..15 are pure X_cont)
        if (wave == 0 && kk + 1 <= 15)
            tdm_load_f32_2d((unsigned)(uintptr_t)fstage[(kk + 1) & 1],
                            Xrow0 + ((kk + 1) * 32 - 10),
                            32, 32, 64, 525);
        if (kk >= 1 && kk <= 15) {
            if (wave == 0) {                    // chunk kk arrived; kk+1 stays in flight
                if (kk < 15) __builtin_amdgcn_s_wait_tensorcnt(1);
                else         __builtin_amdgcn_s_wait_tensorcnt(0);
            }
        } else if (kk == 16) {                  // pad chunk, cooperative
            for (int p = tid; p < 64 * 32; p += 128) {
                int r = p >> 5, k = p & 31, kg = kbase + k;
                cur[p] = (kg < KFEAT) ? Xrow0[(size_t)r * 525 + (kg - 10)] : 0.f;
            }
        }
#else
        if (kk >= 1) {                          // cooperative staging fallback
            if (kk <= 15) {
                for (int p = tid; p < 64 * 32; p += 128) {
                    int r = p >> 5, k = p & 31;
                    cur[p] = Xrow0[(size_t)r * 525 + (kbase + k - 10)];
                }
            } else {
                for (int p = tid; p < 64 * 32; p += 128) {
                    int r = p >> 5, k = p & 31, kg = kbase + k;
                    cur[p] = (kg < KFEAT) ? Xrow0[(size_t)r * 525 + (kg - 10)] : 0.f;
                }
            }
        }
#endif
        __syncthreads();                        // chunk kk visible to all waves
        v16bf a  = make_a_frag_f32lds(&cur[(wave * 16 + nloc) * 32], half);
        v16bf bA = make_b_frag_lds(&wbf[nloc * KPAD],        kbase + half * 16);
        v16bf bB = make_b_frag_lds(&wbf[(16 + nloc) * KPAD], kbase + half * 16);
        acc0 = wmma_bf16(a, bA, acc0);
        acc1 = wmma_bf16(a, bB, acc1);
        __syncthreads();                        // cur free for DMA reuse
    }
    const float bv0 = fc1_b[nloc], bv1 = fc1_b[16 + nloc];
#pragma unroll
    for (int r = 0; r < 8; ++r) {
        int m = wave * 16 + half * 8 + r;
        size_t row = (size_t)n * BATCH + bt0 + m;
        Xb[row * 32 + nloc]      = tobf(fmaxf(acc0[r] + bv0, 0.f));
        Xb[row * 32 + 16 + nloc] = tobf(fmaxf(acc1[r] + bv1, 0.f));
    }
}

// ---------------------------------------------------------------------------
// Kernel 2: fused per-n batched_pred with TDM weight staging.
// grid (32 batch-tiles, 64 n), block 256 (8 waves)
// ---------------------------------------------------------------------------
__global__ void batched_pred_kernel(const float* __restrict__ P,
                                    const bf16* __restrict__ Xb,
                                    float* __restrict__ pred) {
    // manual LDS layout (50,704 B): pstage (f32, TDM dest) overlays hs
    __shared__ char smem[50704];
    bf16*  w1s    = (bf16*) (smem);            // 4096 bf16
    bf16*  w2s    = (bf16*) (smem + 8192);     // 4096 bf16
    float* b1s    = (float*)(smem + 16384);    // 128 f32
    float* b2s    = (float*)(smem + 16896);    // 32 f32
    float* w3s    = (float*)(smem + 17024);    // 32 f32
    float* b3p    = (float*)(smem + 17152);    // 1 f32
    bf16*  hs     = (bf16*) (smem + 17160);    // 16384 bf16
    float* pstage = (float*)(smem + 17160);    // 8385 f32, pre-hs only

    const int tid  = threadIdx.x;
    const int lane = tid & 31, wave = tid >> 5;
    const int half = lane >> 4, nloc = lane & 15;
    const int n    = blockIdx.y;
    const int bt0  = blockIdx.x * 128;
    const float* Pn = P + (size_t)n * ESZ;

#if HAVE_TDM
    if (wave == 0) {
        tdm_load_f32_2d((unsigned)(uintptr_t)pstage, Pn, ESZ, ESZ, 1, ESZ);
        __builtin_amdgcn_s_wait_tensorcnt(0);
    }
    __syncthreads();
    for (int p = tid; p < 4096; p += 256) {
        w1s[p] = tobf(pstage[p]);              // W1 (128,32)
        w2s[p] = tobf(pstage[4224 + p]);       // W2 (32,128)
    }
    if (tid < 128) b1s[tid] = pstage[4096 + tid];
    if (tid < 32)  b2s[tid] = pstage[8320 + tid];
    if (tid >= 32 && tid < 64) w3s[tid - 32] = pstage[8352 + (tid - 32)];
    if (tid == 64) *b3p = pstage[8384];
#else
    for (int p = tid; p < 4096; p += 256) {
        w1s[p] = tobf(Pn[p]);
        w2s[p] = tobf(Pn[4224 + p]);
    }
    if (tid < 128) b1s[tid] = Pn[4096 + tid];
    if (tid < 32)  b2s[tid] = Pn[8320 + tid];
    if (tid >= 32 && tid < 64) w3s[tid - 32] = Pn[8352 + (tid - 32)];
    if (tid == 64) *b3p = Pn[8384];
#endif
    __syncthreads();   // weights converted; pstage region now reusable as hs

    // layer 1: A fragment straight from global bf16 X (K=32, one step)
    const bf16* Xrow = Xb + ((size_t)n * BATCH + bt0 + wave * 16 + nloc) * 32;
    v16bf a1 = make_a_frag(Xrow, 0, half);
#pragma unroll
    for (int ct = 0; ct < 8; ++ct) {
        int col = ct * 16 + nloc;
        v16bf bfr = make_b_frag_lds(&w1s[col * 32], half * 16);
        v8f acc = {};
        acc = wmma_bf16(a1, bfr, acc);
        float bv = b1s[col];
#pragma unroll
        for (int r = 0; r < 8; ++r) {
            int m = wave * 16 + half * 8 + r;             // wave-local rows only
            hs[m * 128 + col] = tobf(fmaxf(acc[r] + bv, 0.f));
        }
    }
    __syncthreads();

    // layer 2: K=128 (4 steps), N=32 (2 col tiles)
    v8f o0 = {}, o1 = {};
    const bf16* hrow = &hs[(wave * 16 + nloc) * 128];
#pragma unroll
    for (int ks = 0; ks < 4; ++ks) {
        v16bf a2  = make_a_frag(hrow, ks * 32, half);
        v16bf bb0 = make_b_frag_lds(&w2s[nloc * 128],        ks * 32 + half * 16);
        v16bf bb1 = make_b_frag_lds(&w2s[(16 + nloc) * 128], ks * 32 + half * 16);
        o0 = wmma_bf16(a2, bb0, o0);
        o1 = wmma_bf16(a2, bb1, o1);
    }

    const float w3a = w3s[nloc], w3b = w3s[16 + nloc];
    const float bva = b2s[nloc], bvb = b2s[16 + nloc];
    float psum[8];
#pragma unroll
    for (int r = 0; r < 8; ++r) {
        float h0 = fmaxf(o0[r] + bva, 0.f);
        float h1 = fmaxf(o1[r] + bvb, 0.f);
        psum[r] = h0 * w3a + h1 * w3b;
    }
#pragma unroll
    for (int m = 1; m < 16; m <<= 1) {
#pragma unroll
        for (int r = 0; r < 8; ++r) psum[r] += __shfl_xor(psum[r], m, 32);
    }
    if (nloc == 0) {
        float b3 = *b3p;
#pragma unroll
        for (int r = 0; r < 8; ++r) {
            int row = wave * 16 + half * 8 + r;
            float y = 1.f / (1.f + __expf(-(psum[r] + b3)));
            pred[(size_t)n * BATCH + bt0 + row] = y;
        }
    }
}

// ---------------------------------------------------------------------------
// Kernel 3: WMMA GEMM  out[m][n] = act(sum_k A[m][k]*W[n][k] + b[n]),  M == 64
// A chunk arrives via TDM (tensor_dim0 = K-kb -> hardware zero-fills tail);
// each B fragment feeds FOUR WMMAs (row tiles). block 128 (4 waves).
// ---------------------------------------------------------------------------
#define GEMM_KC 128
__global__ void gemm_bias_act(const float* __restrict__ A,
                              const float* __restrict__ W,
                              const float* __restrict__ bias,
                              float* __restrict__ out,
                              int N, int K, int do_relu) {
    __shared__ float aT[64 * GEMM_KC];   // 32 KB f32 A chunk
    const int tid  = threadIdx.x;
    const int lane = tid & 31, wave = tid >> 5;
    const int half = lane >> 4, nloc = lane & 15;
    const int col  = blockIdx.x * 64 + wave * 16 + nloc;
    const int ccol = (col < N) ? col : (N - 1);           // safe dummy address
    const float* wrow = W + (size_t)ccol * K;

    v8f acc[4] = {{}, {}, {}, {}};
    for (int kb = 0; kb < K; kb += GEMM_KC) {
        const bool fullK = (kb + GEMM_KC <= K);
#if HAVE_TDM
        if (wave == 0) {
            tdm_load_f32_2d((unsigned)(uintptr_t)aT, A + kb,
                            (unsigned)(K - kb), GEMM_KC, 64, (unsigned)K);
            __builtin_amdgcn_s_wait_tensorcnt(0);
        }
#else
        for (int p = tid; p < 64 * GEMM_KC; p += 128) {
            int r = p >> 7, k = p & (GEMM_KC - 1);
            aT[p] = (kb + k < K) ? A[(size_t)r * K + kb + k] : 0.f;
        }
#endif
        __syncthreads();
        __builtin_prefetch(wrow + kb + GEMM_KC, 0, 1);    // next W chunk hint
#pragma unroll
        for (int kk = 0; kk < GEMM_KC / 32; ++kk) {
            if (kb + kk * 32 < K) {
                int kbb = kb + kk * 32 + half * 16;
                v16bf b = fullK ? make_b_frag_f32_fast(wrow, kbb)
                                : make_b_frag_f32_guard(wrow, kbb, K);
#pragma unroll
                for (int rt = 0; rt < 4; ++rt) {
                    v16bf a = make_a_frag_f32lds(
                        &aT[(rt * 16 + nloc) * GEMM_KC + kk * 32], half);
                    acc[rt] = wmma_bf16(a, b, acc[rt]);
                }
            }
        }
        __syncthreads();
    }
    if (col < N) {
        float bv = bias[col];
#pragma unroll
        for (int rt = 0; rt < 4; ++rt) {
#pragma unroll
            for (int r = 0; r < 8; ++r) {
                int m = rt * 16 + half * 8 + r;
                float v = acc[rt][r] + bv;
                if (do_relu) v = fmaxf(v, 0.f);
                out[(size_t)m * N + col] = v;
            }
        }
    }
}

// ---------------------------------------------------------------------------
// Kernel 4: serial Neural-ODE RK4 scan (latency-bound; one 128-thread block)
// ---------------------------------------------------------------------------
__device__ __forceinline__ void ode_f(const float* vin, float* vout,
                                      const float* W1s, const float* b1s,
                                      const float* W2s, const float* b2s,
                                      float* zs, int tid) {
    float s = b1s[tid];
#pragma unroll 8
    for (int k = 0; k < 32; ++k) s += W1s[tid * 32 + k] * vin[k];
    zs[tid] = tanhf(s);
    __syncthreads();
    if (tid < 32) {
        float s2 = b2s[tid];
#pragma unroll 8
        for (int h = 0; h < 128; ++h) s2 += W2s[tid * 128 + h] * zs[h];
        vout[tid] = s2;
    }
    __syncthreads();
}

__global__ void ode_kernel(const float* __restrict__ e0_in,
                           const float* __restrict__ tarr,
                           const float* __restrict__ W1, const float* __restrict__ B1,
                           const float* __restrict__ W2, const float* __restrict__ B2,
                           float* __restrict__ E_embed) {
    __shared__ float W1s[128 * 32];
    __shared__ float W2s[32 * 128];
    __shared__ float b1s[128], b2s[32];
    __shared__ float ys[32], yt[32], zs[128];
    __shared__ float kk1[32], kk2[32], kk3[32], kk4[32];
    const int tid = threadIdx.x;

    for (int p = tid; p < 4096; p += 128) { W1s[p] = W1[p]; W2s[p] = W2[p]; }
    if (tid < 128) b1s[tid] = B1[tid];
    if (tid < 32) { b2s[tid] = B2[tid]; ys[tid] = e0_in[tid]; E_embed[tid] = e0_in[tid]; }
    __syncthreads();

    for (int i = 1; i < NT; ++i) {
        float dt = tarr[i] - tarr[i - 1];
        float hh = dt * 0.2f;                  // dt / N_SUB
        for (int s = 0; s < 5; ++s) {
            ode_f(ys, kk1, W1s, b1s, W2s, b2s, zs, tid);
            if (tid < 32) yt[tid] = ys[tid] + 0.5f * hh * kk1[tid];
            __syncthreads();
            ode_f(yt, kk2, W1s, b1s, W2s, b2s, zs, tid);
            if (tid < 32) yt[tid] = ys[tid] + 0.5f * hh * kk2[tid];
            __syncthreads();
            ode_f(yt, kk3, W1s, b1s, W2s, b2s, zs, tid);
            if (tid < 32) yt[tid] = ys[tid] + hh * kk3[tid];
            __syncthreads();
            ode_f(yt, kk4, W1s, b1s, W2s, b2s, zs, tid);
            if (tid < 32)
                ys[tid] += (hh * (1.f / 6.f)) *
                           (kk1[tid] + 2.f * kk2[tid] + 2.f * kk3[tid] + kk4[tid]);
            __syncthreads();
        }
        if (tid < 32) E_embed[i * 32 + tid] = ys[tid];
        __syncthreads();
    }
}

// ---------------------------------------------------------------------------
// Kernel 5: param_stack -> init_param passthrough (float4 copy)
// ---------------------------------------------------------------------------
__global__ void copy_f32v4(const float4* __restrict__ src, float4* __restrict__ dst, int n4) {
    int i = blockIdx.x * blockDim.x + threadIdx.x;
    if (i < n4) dst[i] = src[i];
}

// ---------------------------------------------------------------------------
// Host orchestration
// ---------------------------------------------------------------------------
extern "C" void kernel_launch(void* const* d_in, const int* in_sizes, int n_in,
                              void* d_out, int out_size, void* d_ws, size_t ws_size,
                              hipStream_t stream) {
    const int*   disc_idx   = (const int*)  d_in[0];
    const float* X_cont     = (const float*)d_in[1];
    const float* t_arr      = (const float*)d_in[2];
    const float* param_stk  = (const float*)d_in[3];
    const float* embedding  = (const float*)d_in[4];
    const float* fc1_w      = (const float*)d_in[5];
    const float* fc1_b      = (const float*)d_in[6];
    const float* enc_w1 = (const float*)d_in[7];  const float* enc_b1 = (const float*)d_in[8];
    const float* enc_w2 = (const float*)d_in[9];  const float* enc_b2 = (const float*)d_in[10];
    const float* enc_w3 = (const float*)d_in[11]; const float* enc_b3 = (const float*)d_in[12];
    const float* enc_w4 = (const float*)d_in[13]; const float* enc_b4 = (const float*)d_in[14];
    const float* dec_w1 = (const float*)d_in[15]; const float* dec_b1 = (const float*)d_in[16];
    const float* dec_w2 = (const float*)d_in[17]; const float* dec_b2 = (const float*)d_in[18];
    const float* dec_w3 = (const float*)d_in[19]; const float* dec_b3 = (const float*)d_in[20];
    const float* dec_w4 = (const float*)d_in[21]; const float* dec_b4 = (const float*)d_in[22];
    const float* ode_w1 = (const float*)d_in[23]; const float* ode_b1 = (const float*)d_in[24];
    const float* ode_w2 = (const float*)d_in[25]; const float* ode_b2 = (const float*)d_in[26];
    (void)in_sizes; (void)n_in; (void)out_size; (void)ws_size;

    // output layout (floats): E, pred, init_param, init_pred, init_embed, E_embed, init_debed
    float* out          = (float*)d_out;
    float* E_out        = out;                    // 64*8385 = 536640
    float* pred_out     = out + 536640;           // 262144
    float* initp_out    = out + 798784;           // 536640
    float* initpred_out = out + 1335424;          // 262144
    float* inite_out    = out + 1597568;          // 2048
    float* Eemb_out     = out + 1599616;          // 2048
    float* initdeb_out  = out + 1601664;          // 536640

    // workspace: Xb (bf16 X, 16.78 MB) + f32 temps
    bf16* Xb  = (bf16*)d_ws;
    char* wsb = (char*)d_ws;
    float* t1 = (float*)(wsb + 16777216);         // 64*1024 f32
    float* t2 = (float*)(wsb + 17039360);         // 64*512  f32
    float* t3 = (float*)(wsb + 17170432);         // 64*128  f32

    auto gemm = [&](const float* A, const float* W, const float* b, float* o,
                    int N, int K, int relu) {
        gemm_bias_act<<<dim3((N + 63) / 64), 128, 0, stream>>>(A, W, b, o, N, K, relu);
    };

    // init_param passthrough
    copy_f32v4<<<(134160 + 255) / 256, 256, 0, stream>>>(
        (const float4*)param_stk, (float4*)initp_out, 134160);

    // X = relu(fc1(concat(emb[disc_idx], X_cont)))  -> bf16 workspace
    fc1_kernel<<<dim3(64, NT), 128, 0, stream>>>(disc_idx, X_cont, embedding,
                                                 fc1_w, fc1_b, Xb);

    // init_pred = batched_pred(param_stack, X)
    batched_pred_kernel<<<dim3(32, NT), 256, 0, stream>>>(param_stk, Xb, initpred_out);

    // init_embed = enc_mlp4(param_stack)
    gemm(param_stk, enc_w1, enc_b1, t1, 1024, ESZ, 1);
    gemm(t1, enc_w2, enc_b2, t2, 512, 1024, 1);
    gemm(t2, enc_w3, enc_b3, t3, 128, 512, 1);
    gemm(t3, enc_w4, enc_b4, inite_out, 32, 128, 0);

    // E_embed = RK4 scan from e0 = init_embed[0]
    ode_kernel<<<1, 128, 0, stream>>>(inite_out, t_arr, ode_w1, ode_b1,
                                      ode_w2, ode_b2, Eemb_out);

    // init_debed = dec_mlp4(init_embed)
    gemm(inite_out, dec_w1, dec_b1, t3, 128, 32, 1);
    gemm(t3, dec_w2, dec_b2, t2, 512, 128, 1);
    gemm(t2, dec_w3, dec_b3, t1, 1024, 512, 1);
    gemm(t1, dec_w4, dec_b4, initdeb_out, ESZ, 1024, 0);

    // E = dec_mlp4(E_embed)
    gemm(Eemb_out, dec_w1, dec_b1, t3, 128, 32, 1);
    gemm(t3, dec_w2, dec_b2, t2, 512, 128, 1);
    gemm(t2, dec_w3, dec_b3, t1, 1024, 512, 1);
    gemm(t1, dec_w4, dec_b4, E_out, ESZ, 1024, 0);

    // pred = batched_pred(E, X)
    batched_pred_kernel<<<dim3(32, NT), 256, 0, stream>>>(E_out, Xb, pred_out);
}
